// ExpertBank_21406117003479
// MI455X (gfx1250) — matrix operand order repaired and checked by reading, not compiled, not run on hardware
//
#include <hip/hip_runtime.h>
#include <hip/hip_bf16.h>
#include <math.h>

// ---------------------------------------------------------------------------
// ExpertBank: out[t,e,:] = gelu( xhat[t,:] @ (g[e]*W1[e]) + b1'[e] ) @ W2[e] + b2[e]
//   xhat = layernorm(tokens) (no affine), b1'[e] = b1[e] + ln_b[e] @ W1[e]
// H=512, F=1024, E=16, T=8192. f16 WMMA (v_wmma_f32_16x16x32_f16), f32 accum.
// ---------------------------------------------------------------------------

typedef __attribute__((ext_vector_type(16))) _Float16 v16h;
typedef __attribute__((ext_vector_type(8)))  float    v8f;
typedef __attribute__((ext_vector_type(4)))  unsigned int u32x4;

union V16H { v16h h; u32x4 q[2]; };

#define H_DIM 512
#define F_DIM 1024
#define E_CNT 16
#define T_CNT 8192
#define MT    64          // token tile per workgroup
#define XPITCH 1040       // 512 f16 = 1024B + 16B pad (bank stagger)
#define HPITCH 2064       // 1024 f16 = 2048B + 16B pad

// ws layout (bytes)
#define WS_XHAT 0                         // 8192*512*2            = 8,388,608
#define WS_W1F  8388608                   // 16*512*1024*2         = 16,777,216
#define WS_B1E  25165824                  // 16*1024*4             = 65,536
#define WS_W2F  25231360                  // 16*1024*512*2         = 16,777,216
// total required ws: ~42,008,576 bytes

// ---------------- Prepass 1: LayerNorm -> xhat (f16, row major) -------------
__global__ __launch_bounds__(256) void k_ln(const float* __restrict__ tokens,
                                            _Float16* __restrict__ xhat) {
    const int wid  = threadIdx.x >> 5;
    const int lane = threadIdx.x & 31;
    const int t    = blockIdx.x * 8 + wid;
    const float* row = tokens + (size_t)t * H_DIM;
    float v[16];
    float s = 0.f, ss = 0.f;
#pragma unroll
    for (int j = 0; j < 16; ++j) {
        float x = row[j * 32 + lane];
        v[j] = x; s += x; ss += x * x;
    }
#pragma unroll
    for (int off = 16; off > 0; off >>= 1) {
        s  += __shfl_xor(s,  off, 32);
        ss += __shfl_xor(ss, off, 32);
    }
    const float mu  = s * (1.f / H_DIM);
    const float var = ss * (1.f / H_DIM) - mu * mu;
    const float rs  = rsqrtf(var + 1e-5f);
    _Float16* orow = xhat + (size_t)t * H_DIM;
#pragma unroll
    for (int j = 0; j < 16; ++j)
        orow[j * 32 + lane] = (_Float16)((v[j] - mu) * rs);
}

// ------- Prepass 2: W1' = diag(ln_g)*W1 -> f16 B-fragment layout ------------
// frag(e,ks,nt): 32(K)x16(N) tile, K = ks*32.., N = nt*16..
// per-lane 32B: lane L holds N = nt*16 + L%16, K = ks*32 + (L/16)*16 + {0..15}
__global__ __launch_bounds__(256) void k_w1(const float* __restrict__ W1,
                                            const float* __restrict__ lng,
                                            _Float16* __restrict__ out) {
    const int tid  = blockIdx.x * 256 + threadIdx.x;   // 524288 threads
    const int lane = tid & 31;
    const int frag = tid >> 5;                         // e*1024 + ks*64 + nt
    const int nt = frag & 63;
    const int ks = (frag >> 6) & 15;
    const int e  = frag >> 10;
    const int n     = nt * 16 + (lane & 15);
    const int kbase = ks * 32 + ((lane >> 4) << 4);
    const float* Wp = W1 + (size_t)e * H_DIM * F_DIM;
    const float* gp = lng + e * H_DIM;
    _Float16* op = out + (size_t)frag * 512 + lane * 16;   // frag = 1024B
#pragma unroll
    for (int r = 0; r < 8; ++r) {
        const int k0 = kbase + 2 * r;
        op[2 * r]     = (_Float16)(Wp[(size_t)k0 * F_DIM + n]       * gp[k0]);
        op[2 * r + 1] = (_Float16)(Wp[(size_t)(k0 + 1) * F_DIM + n] * gp[k0 + 1]);
    }
}

// ------- Prepass 3: W2 -> f16 B-fragment layout (K in F, N in H) ------------
__global__ __launch_bounds__(256) void k_w2(const float* __restrict__ W2,
                                            _Float16* __restrict__ out) {
    const int tid  = blockIdx.x * 256 + threadIdx.x;   // 524288 threads
    const int lane = tid & 31;
    const int frag = tid >> 5;                         // e*1024 + ks*32 + nt
    const int nt = frag & 31;
    const int ks = (frag >> 5) & 31;
    const int e  = frag >> 10;
    const int n     = nt * 16 + (lane & 15);
    const int kbase = ks * 32 + ((lane >> 4) << 4);
    const float* Wp = W2 + (size_t)e * F_DIM * H_DIM;
    _Float16* op = out + (size_t)frag * 512 + lane * 16;
#pragma unroll
    for (int r = 0; r < 8; ++r) {
        const int k0 = kbase + 2 * r;
        op[2 * r]     = (_Float16)Wp[(size_t)k0 * H_DIM + n];
        op[2 * r + 1] = (_Float16)Wp[(size_t)(k0 + 1) * H_DIM + n];
    }
}

// ------- Prepass 4: b1'[e,f] = b1[e,f] + sum_h ln_b[e,h]*W1[e,h,f] ----------
__global__ __launch_bounds__(256) void k_b1(const float* __restrict__ W1,
                                            const float* __restrict__ lnb,
                                            const float* __restrict__ b1,
                                            float* __restrict__ out) {
    const int tid = blockIdx.x * 256 + threadIdx.x;    // 16384
    const int e = tid >> 10, f = tid & 1023;
    const float* Wp = W1 + (size_t)e * H_DIM * F_DIM + f;
    const float* bp = lnb + e * H_DIM;
    float acc = b1[tid];
    for (int h = 0; h < H_DIM; ++h)
        acc += bp[h] * Wp[(size_t)h * F_DIM];
    out[tid] = acc;
}

// ---------------- Main kernel: double GEMM via WMMA -------------------------
__global__ __launch_bounds__(256) void k_moe(const _Float16* __restrict__ xhat,
                                             const _Float16* __restrict__ w1f,
                                             const float* __restrict__ b1e,
                                             const _Float16* __restrict__ w2f,
                                             const float* __restrict__ b2,
                                             float* __restrict__ out) {
    extern __shared__ char smem[];
    char* xA = smem;                       // MT rows * XPITCH
    char* hA = smem + MT * XPITCH;         // MT rows * HPITCH

    const int tid  = threadIdx.x;
    const int lane = tid & 31;
    const int w    = tid >> 5;             // wave 0..7
    const int e    = blockIdx.y;
    const int t0   = blockIdx.x * MT;

    // ---- Phase 0: stage xhat tile (64 x 512 f16) into LDS, coalesced ----
    {
        const u32x4* src = (const u32x4*)(xhat + (size_t)t0 * H_DIM);
        for (int idx = tid; idx < 64 * 64; idx += 256) {
            const int row = idx >> 6, c = idx & 63;
            *(u32x4*)(xA + row * XPITCH + c * 16) = src[row * 64 + c];
        }
    }
    __syncthreads();

    const int m0 = (w >> 1) * 16;          // token sub-tile base
    const int nh = w & 1;                  // N half
    const int lm = lane & 15;
    const int lh = lane >> 4;              // 0/1

    // ---- Phase 1: h = gelu(xhat @ W1' + b1') ----
    for (int c = 0; c < 4; ++c) {
        const int nt0 = nh * 32 + c * 8;
        v8f acc[8] = {};
        for (int ks = 0; ks < 16; ++ks) {
            V16H a;
            const char* ap = xA + (m0 + lm) * XPITCH + ks * 64 + lh * 32;
            a.q[0] = *(const u32x4*)ap;
            a.q[1] = *(const u32x4*)(ap + 16);
#pragma unroll
            for (int j = 0; j < 8; ++j) {
                const v16h b = *(const v16h*)(w1f +
                    (size_t)((e * 16 + ks) * 64 + nt0 + j) * 512 + lane * 16);
                acc[j] = __builtin_amdgcn_wmma_f32_16x16x32_f16(
                    false, a.h, false, b, (short)0, acc[j], false, false);
            }
        }
#pragma unroll
        for (int j = 0; j < 8; ++j) {
            const int col  = (nt0 + j) * 16 + lm;
            const float bias = b1e[e * F_DIM + col];
#pragma unroll
            for (int r = 0; r < 8; ++r) {
                const int mrow = m0 + r + lh * 8;
                const float v  = acc[j][r] + bias;
                const float gl = 0.5f * v * (1.f + erff(v * 0.7071067811865475f));
                *(_Float16*)(hA + mrow * HPITCH + col * 2) = (_Float16)gl;
            }
        }
    }
    __syncthreads();

    // ---- Phase 2: out = h @ W2' + b2 ----
    for (int c = 0; c < 2; ++c) {
        const int nt0 = nh * 16 + c * 8;
        v8f acc[8] = {};
        for (int ks = 0; ks < 32; ++ks) {
            V16H a;
            const char* ap = hA + (m0 + lm) * HPITCH + ks * 64 + lh * 32;
            a.q[0] = *(const u32x4*)ap;
            a.q[1] = *(const u32x4*)(ap + 16);
#pragma unroll
            for (int j = 0; j < 8; ++j) {
                const v16h b = *(const v16h*)(w2f +
                    (size_t)((e * 32 + ks) * 32 + nt0 + j) * 512 + lane * 16);
                acc[j] = __builtin_amdgcn_wmma_f32_16x16x32_f16(
                    false, a.h, false, b, (short)0, acc[j], false, false);
            }
        }
#pragma unroll
        for (int j = 0; j < 8; ++j) {
            const int col  = (nt0 + j) * 16 + lm;
            const float bias = b2[e * H_DIM + col];
#pragma unroll
            for (int r = 0; r < 8; ++r) {
                const size_t token = (size_t)(t0 + m0 + r + lh * 8);
                out[(token * E_CNT + e) * H_DIM + col] = acc[j][r] + bias;
            }
        }
    }
}

// ---------------------------------------------------------------------------
extern "C" void kernel_launch(void* const* d_in, const int* in_sizes, int n_in,
                              void* d_out, int out_size, void* d_ws, size_t ws_size,
                              hipStream_t stream) {
    const float* tokens = (const float*)d_in[0];
    const float* ln_g   = (const float*)d_in[1];
    const float* ln_b   = (const float*)d_in[2];
    const float* W1     = (const float*)d_in[3];
    const float* b1     = (const float*)d_in[4];
    const float* W2     = (const float*)d_in[5];
    const float* b2     = (const float*)d_in[6];
    float* out = (float*)d_out;
    char*  ws  = (char*)d_ws;

    _Float16* xhat = (_Float16*)(ws + WS_XHAT);
    _Float16* w1f  = (_Float16*)(ws + WS_W1F);
    float*    b1e  = (float*)   (ws + WS_B1E);
    _Float16* w2f  = (_Float16*)(ws + WS_W2F);

    k_ln<<<T_CNT / 8, 256, 0, stream>>>(tokens, xhat);
    k_w1<<<2048, 256, 0, stream>>>(W1, ln_g, w1f);
    k_w2<<<2048, 256, 0, stream>>>(W2, w2f);
    k_b1<<<64, 256, 0, stream>>>(W1, ln_b, b1, b1e);

    dim3 grid(T_CNT / MT, E_CNT);
    const size_t lds = (size_t)MT * XPITCH + (size_t)MT * HPITCH;  // 198,656 B
    k_moe<<<grid, 256, lds, stream>>>(xhat, w1f, b1e, w2f, b2, out);
}
